// SpinConvLayer_12635793785394
// MI455X (gfx1250) — matrix operand order repaired and checked by reading, not compiled, not run on hardware
//
#include <hip/hip_runtime.h>
#include <hip/hip_bf16.h>

#define N_ATOM 16384
#define N_EDGE 262144
#define PI_F 3.14159265358979f

typedef __attribute__((ext_vector_type(16))) _Float16 v16h;
typedef __attribute__((ext_vector_type(8)))  float    v8f;

union Frag {
    v16h h;
    unsigned u[8];
};

__device__ __forceinline__ int kidxA(int v, int hi) {
    // 16-bit A-matrix 16x32 layout: lanes 0-15 -> K {0..7,16..23}, lanes 16-31 -> +8
    return ((v < 4) ? (2 * v) : (16 + 2 * (v - 4))) + hi;
}
__device__ __forceinline__ int kidxB(int v, int hi2) {
    // 16-bit B-matrix 32x16 layout: lanes 0-15 -> K {0..7,16..23}, lanes 16-31 -> +8
    return ((v < 4) ? (2 * v) : (16 + 2 * (v - 4))) + (hi2 >> 1);
}
__device__ __forceinline__ unsigned packh(float a, float b) {
    union { _Float16 h[2]; unsigned u; } t;
    t.h[0] = (_Float16)a;
    t.h[1] = (_Float16)b;
    return t.u;
}
__device__ __forceinline__ v8f wmma_f16(v16h a, v16h b, v8f c) {
    return __builtin_amdgcn_wmma_f32_16x16x32_f16(false, a, false, b, (short)0, c, false, false);
}

// ---------------------------------------------------------------------------
// Kernel 1: per-node spin radial basis -> sb[way][node][96] in workspace
// ---------------------------------------------------------------------------
__global__ void spinconv_sb_kernel(const float* __restrict__ spin,
                                   const float* __restrict__ srbf_w,
                                   const float* __restrict__ srbf_b,
                                   float* __restrict__ sb) {
    int n = blockIdx.x;
    int c = threadIdx.x;                      // 0..95
    float sx = spin[n * 3 + 0];
    float sy = spin[n * 3 + 1];
    float sz = spin[n * 3 + 2];
    float mi = sqrtf(sx * sx + sy * sy + sz * sz + 1e-12f);
    float g[8];
#pragma unroll
    for (int k = 0; k < 8; ++k) {
        float d = mi - (3.0f / 7.0f) * (float)k;
        g[k] = __expf(-4.0f * d * d);
    }
#pragma unroll
    for (int way = 0; way < 3; ++way) {
        float acc = srbf_b[way * 96 + c];
#pragma unroll
        for (int k = 0; k < 8; ++k)
            acc += g[k] * srbf_w[way * 768 + k * 96 + c];
        sb[(size_t)way * (N_ATOM * 96) + (size_t)n * 96 + c] = acc;
    }
}

// ---------------------------------------------------------------------------
// Kernel 2: fused edge kernel. 64 threads = 2 waves, each wave owns 16 edges.
// ---------------------------------------------------------------------------
__global__ void __launch_bounds__(64)
spinconv_edge_kernel(const float* __restrict__ node0,
                     const float* __restrict__ node1,
                     const float* __restrict__ node2,
                     const float* __restrict__ coord,
                     const float* __restrict__ spin,
                     const int* __restrict__ idx_i,
                     const int* __restrict__ idx_j,
                     const float* __restrict__ rbf_w,
                     const float* __restrict__ rbf_b,
                     const float* __restrict__ U_w,
                     const float* __restrict__ V_w,
                     const float* __restrict__ sb,
                     float* __restrict__ out0,
                     float* __restrict__ out1,
                     float* __restrict__ out2) {
    // Block-shared weight staging (f16, transposed to [way][o][k] for B-frags)
    __shared__ _Float16 Uh[3 * 32 * 32];     //  6 KB
    __shared__ _Float16 Vh[3 * 32 * 96];     // 18 KB
    __shared__ float    rbfwL[3 * 8 * 96];   //  9 KB
    __shared__ float    rbfbL[3 * 96];
    // Per-wave regions
    __shared__ float    uL[2][16 * 3];
    __shared__ float    chebL[2][16 * 3];
    __shared__ float    rbfL[2][16 * 8];
    __shared__ int      idxL[2][16 * 2];
    __shared__ _Float16 xhL[2][3 * 16 * 96]; // 9 KB per wave

    const int tid  = threadIdx.x;
    const int w    = tid >> 5;
    const int lane = tid & 31;
    const int e0   = (blockIdx.x * 2 + w) * 16;

    // ---- Phase 0: stage weights to LDS (transpose + f32->f16) ----
    for (int t = tid; t < 3072; t += 64) {          // Uh[way][o][k] = U_w[way][k][o]
        int way = t >> 10, r = t & 1023, o = r >> 5, k = r & 31;
        Uh[t] = (_Float16)U_w[way * 1024 + k * 32 + o];
    }
    for (int t = tid; t < 9216; t += 64) {          // Vh[way][o][k] = V_w[way][k][o]
        int way = t / 3072, r = t % 3072, o = r / 96, k = r % 96;
        Vh[t] = (_Float16)V_w[way * 3072 + k * 32 + o];
    }
    for (int t = tid; t < 2304; t += 64) rbfwL[t] = rbf_w[t];
    for (int t = tid; t < 288;  t += 64) rbfbL[t] = rbf_b[t];

    // ---- Phase 1: per-edge scalars (lanes 0..15, one edge each) ----
    if (lane < 16) {
        int e  = e0 + lane;
        int ii = idx_i[e];
        int jj = idx_j[e];
        idxL[w][lane * 2 + 0] = ii;
        idxL[w][lane * 2 + 1] = jj;
        float rx = coord[jj * 3 + 0] - coord[ii * 3 + 0];
        float ry = coord[jj * 3 + 1] - coord[ii * 3 + 1];
        float rz = coord[jj * 3 + 2] - coord[ii * 3 + 2];
        float d  = sqrtf(rx * rx + ry * ry + rz * rz + 1e-12f);
        float inv = 1.0f / d;
        uL[w][lane * 3 + 0] = rx * inv;
        uL[w][lane * 3 + 1] = ry * inv;
        uL[w][lane * 3 + 2] = rz * inv;
        float fc = 0.5f * (__cosf(PI_F * fminf(d * 0.2f, 1.0f)) + 1.0f);
#pragma unroll
        for (int k = 0; k < 8; ++k) {
            float dd = d - (5.0f / 7.0f) * (float)k;
            rbfL[w][lane * 8 + k] = __expf(-4.0f * dd * dd) * fc;
        }
        float aix = spin[ii * 3 + 0], aiy = spin[ii * 3 + 1], aiz = spin[ii * 3 + 2];
        float ajx = spin[jj * 3 + 0], ajy = spin[jj * 3 + 1], ajz = spin[jj * 3 + 2];
        float mi = sqrtf(aix * aix + aiy * aiy + aiz * aiz + 1e-12f);
        float mj = sqrtf(ajx * ajx + ajy * ajy + ajz * ajz + 1e-12f);
        float sij = (aix * ajx + aiy * ajy + aiz * ajz) / (mi * mj);
        chebL[w][lane * 3 + 0] = 1.0f;
        chebL[w][lane * 3 + 1] = sij;
        chebL[w][lane * 3 + 2] = 2.0f * sij * sij - 1.0f;
    }
    __syncthreads();

    // ---- Phase 2: xixi (A operand for V GEMM) -> LDS f16 [way][m][96] ----
    for (int t = lane; t < 4608; t += 32) {
        int way = t / 1536;
        int r   = t - way * 1536;
        int m   = r / 96;
        int c   = r - m * 96;
        int tb  = c >> 5;
        float acc = rbfbL[way * 96 + c];
#pragma unroll
        for (int k = 0; k < 8; ++k)
            acc += rbfL[w][m * 8 + k] * rbfwL[way * 768 + k * 96 + c];
        int ii = idxL[w][m * 2 + 0];
        int jj = idxL[w][m * 2 + 1];
        float s = sb[(size_t)way * (N_ATOM * 96) + (size_t)ii * 96 + c] *
                  sb[(size_t)way * (N_ATOM * 96) + (size_t)jj * 96 + c];
        xhL[w][way * 1536 + m * 96 + c] = (_Float16)(acc * s * chebL[w][m * 3 + tb]);
    }
    __syncthreads();

    const int hi  = (lane >= 16) ? 8 : 0;    // A-frag K half-offset
    const int hi2 = (lane >= 16) ? 16 : 0;   // B-frag K half-offset
    const int mA  = lane & 15;               // A-frag row (edge within tile)
    const int jm  = idxL[w][mA * 2 + 1];     // gather source node for this row
    const float* base0 = node0 + (size_t)jm * 32;
    const float* base1 = node1 + (size_t)jm * 96;
    const float* base2 = node2 + (size_t)jm * 288;
    const unsigned* Up = (const unsigned*)Uh;
    const unsigned* Vp = (const unsigned*)Vh;
    const unsigned* Xp = (const unsigned*)&xhL[w][0];

    // ---- Phase 4: two 16-channel output tiles ----
#pragma unroll
    for (int nt = 0; nt < 2; ++nt) {
        const int n = (lane & 15) + nt * 16; // out-channel column for B / D frags
        v8f zero = {};

        // B fragments for U (one per way, from LDS)
        Frag bu[3];
#pragma unroll
        for (int way = 0; way < 3; ++way)
#pragma unroll
            for (int v = 0; v < 8; ++v) {
                int k = kidxB(v, hi2);
                bu[way].u[v] = Up[(way * 1024 + n * 32 + k) >> 1];
            }

        // 13 U GEMMs: A gathered straight from global (L2-resident)
        v8f accU[13];
        {
            Frag a;
#pragma unroll
            for (int v = 0; v < 8; ++v) {
                int k = kidxA(v, hi);
                a.u[v] = packh(base0[k], base0[k + 1]);
            }
            accU[0] = wmma_f16(a.h, bu[0].h, zero);
        }
#pragma unroll
        for (int s = 0; s < 3; ++s) {
            Frag a;
#pragma unroll
            for (int v = 0; v < 8; ++v) {
                int k = kidxA(v, hi);
                a.u[v] = packh(base1[k * 3 + s], base1[(k + 1) * 3 + s]);
            }
            accU[1 + s] = wmma_f16(a.h, bu[1].h, zero);
        }
#pragma unroll
        for (int s = 0; s < 9; ++s) {
            Frag a;
#pragma unroll
            for (int v = 0; v < 8; ++v) {
                int k = kidxA(v, hi);
                a.u[v] = packh(base2[k * 9 + s], base2[(k + 1) * 9 + s]);
            }
            accU[4 + s] = wmma_f16(a.h, bu[2].h, zero);
        }

        // 3 V GEMMs (K=96 -> 3 WMMA steps each), A/B from LDS
        v8f accV[3];
#pragma unroll
        for (int way = 0; way < 3; ++way) {
            v8f acc = {};
#pragma unroll
            for (int kb = 0; kb < 3; ++kb) {
                Frag a, b;
#pragma unroll
                for (int v = 0; v < 8; ++v) {
                    int k = kidxA(v, hi);
                    a.u[v] = Xp[(way * 1536 + mA * 96 + kb * 32 + k) >> 1];
                }
#pragma unroll
                for (int v = 0; v < 8; ++v) {
                    int k = kidxB(v, hi2);
                    b.u[v] = Vp[(way * 3072 + n * 96 + kb * 32 + k) >> 1];
                }
                acc = wmma_f16(a.h, b.h, acc);
            }
            accV[way] = acc;
        }

        // Fused tensor-product epilogue + stores.
        // D-frag element (v) in this lane is edge m = v + 8*(lane>=16), channel o = n.
#pragma unroll
        for (int v = 0; v < 8; ++v) {
            int m = v + ((lane >= 16) ? 8 : 0);
            float ux = uL[w][m * 3 + 0];
            float uy = uL[w][m * 3 + 1];
            float uz = uL[w][m * 3 + 2];
            float uu[3] = {ux, uy, uz};
            float n0 = accU[0][v];
            float n1[3], n2[9];
#pragma unroll
            for (int s = 0; s < 3; ++s) n1[s] = accU[1 + s][v];
#pragma unroll
            for (int s = 0; s < 9; ++s) n2[s] = accU[4 + s][v];
            float r0 = accV[0][v], r1 = accV[1][v], r2 = accV[2][v];

            float d1 = n1[0] * ux + n1[1] * uy + n1[2] * uz;
            float n2u[3];
#pragma unroll
            for (int i = 0; i < 3; ++i)
                n2u[i] = n2[3 * i] * ux + n2[3 * i + 1] * uy + n2[3 * i + 2] * uz;
            float quad = n2u[0] * ux + n2u[1] * uy + n2u[2] * uz;

            size_t eo = (size_t)(e0 + m) * 32 + (size_t)n;
            out0[eo] = n0 * r0 + r1 * d1 + r2 * quad;
            size_t b1 = eo * 3;
#pragma unroll
            for (int i = 0; i < 3; ++i)
                out1[b1 + i] = r1 * uu[i] * n0 + r0 * n1[i] + r2 * uu[i] * d1 + r1 * n2u[i];
            size_t b2 = eo * 9;
#pragma unroll
            for (int i = 0; i < 3; ++i)
#pragma unroll
                for (int j = 0; j < 3; ++j)
                    out2[b2 + 3 * i + j] = r2 * uu[i] * uu[j] * n0 + r1 * n1[i] * uu[j] +
                                           r0 * n2[3 * i + j] + r2 * uu[j] * n2u[i];
        }
    }
}

extern "C" void kernel_launch(void* const* d_in, const int* in_sizes, int n_in,
                              void* d_out, int out_size, void* d_ws, size_t ws_size,
                              hipStream_t stream) {
    const float* node0  = (const float*)d_in[0];
    const float* node1  = (const float*)d_in[1];
    const float* node2  = (const float*)d_in[2];
    const float* coord  = (const float*)d_in[3];
    const float* spin   = (const float*)d_in[4];
    const int*   idx_i  = (const int*)d_in[5];
    const int*   idx_j  = (const int*)d_in[6];
    const float* rbf_w  = (const float*)d_in[7];
    const float* rbf_b  = (const float*)d_in[8];
    const float* srbf_w = (const float*)d_in[9];
    const float* srbf_b = (const float*)d_in[10];
    const float* U_w    = (const float*)d_in[11];
    const float* V_w    = (const float*)d_in[12];

    float* out  = (float*)d_out;
    float* out0 = out;                           // [E,32]
    float* out1 = out + (size_t)N_EDGE * 32;     // [E,32,3]
    float* out2 = out + (size_t)N_EDGE * 32 * 4; // [E,32,9]
    float* sb   = (float*)d_ws;                  // [3][N_ATOM][96] f32

    spinconv_sb_kernel<<<N_ATOM, 96, 0, stream>>>(spin, srbf_w, srbf_b, sb);
    spinconv_edge_kernel<<<N_EDGE / 32, 64, 0, stream>>>(
        node0, node1, node2, coord, spin, idx_i, idx_j,
        rbf_w, rbf_b, U_w, V_w, sb, out0, out1, out2);
}
